// FastSoftmaxAttention_21775484191371
// MI455X (gfx1250) — matrix operand order repaired and checked by reading, not compiled, not run on hardware
//
#include <hip/hip_runtime.h>
#include <stdint.h>

// Problem dims
#define BB 16
#define QQ 4096
#define KKDIM 4096
#define DDIM 256
#define DVDIM 256
#define MMDIM 256          // feature count m == D
#define CPAD 272           // c = DV+1 = 257 padded to 17 tiles of 16
#define NSPLIT 4
#define KCHUNK (KKDIM / NSPLIT)
#define NCBLK 3            // 3 blocks x 8 waves cover 17 c-tiles
#define CTILES 17
#define LDSPITCH 520u      // 512B row + 8B TDM pad -> 8B-aligned rows, bank shift 2/row

typedef unsigned int u32;
typedef unsigned short u16;
typedef unsigned long long u64;

typedef __attribute__((ext_vector_type(16))) __bf16 v16bf;
typedef __attribute__((ext_vector_type(8)))  float  v8f;
typedef __attribute__((ext_vector_type(4)))  unsigned int v4u;
typedef __attribute__((ext_vector_type(8)))  int v8i;
typedef __attribute__((ext_vector_type(4)))  int v4i;

#ifndef __has_builtin
#define __has_builtin(x) 0
#endif
#if __has_builtin(__builtin_amdgcn_tensor_load_to_lds) && __has_builtin(__builtin_amdgcn_s_wait_tensorcnt)
#define USE_TDM 1
#else
#define USE_TDM 0
#endif

union Frag {
  u32   w[8];
  uint2 d[4];
  uint4 q[2];
  v16bf v;
};

struct ARaw { float f[16]; };

__device__ __forceinline__ u16 f2bf(float f) {        // native f32->bf16 cvt
  return __builtin_bit_cast(u16, (__bf16)f);
}
__device__ __forceinline__ u32 pack2(float lo, float hi) {
  return (u32)f2bf(lo) | ((u32)f2bf(hi) << 16);
}
__device__ __forceinline__ v8f vzero() {
  v8f z;
#pragma unroll
  for (int i = 0; i < 8; ++i) z[i] = 0.0f;
  return z;
}
__device__ __forceinline__ v8f wmma_bf16(const Frag& a, const Frag& b, v8f c) {
  return __builtin_amdgcn_wmma_f32_16x16x32_bf16(false, a.v, false, b.v,
                                                 (short)0, c, false, false);
}

// Load one B fragment (2 runs of 8 bf16, 16 elements apart) from padded LDS (8B-aligned rows).
__device__ __forceinline__ void lds_bfrag(Frag& f, const char* smem, int row, u32 bofs) {
  const uint2* lp = (const uint2*)(smem + (u32)row * LDSPITCH + bofs);
#pragma unroll
  for (int j = 0; j < 2; ++j) { f.d[j] = lp[j]; f.d[2 + j] = lp[4 + j]; }
}

#if USE_TDM
// Build a 2D bf16 D# per CDNA5 ISA 8.3/8.4 and issue TENSOR_LOAD_TO_LDS.
// pad_interval=6 (64x8B = 512B), pad_amount=1 (2 DWORDs) -> 520B LDS pitch.
__device__ __forceinline__ void tdm_load_2d_bf16(u32 lds_byte_off, const void* src,
                                                 u32 rows, u32 row_elems, u32 stride_elems) {
  u64 ga = (u64)(uintptr_t)src;
  v4u g0;
  g0[0] = 1u;                                                // count=1
  g0[1] = lds_byte_off;                                      // lds_addr
  g0[2] = (u32)ga;                                           // global_addr[31:0]
  g0[3] = (u32)((ga >> 32) & 0x01FFFFFFu) | (2u << 30);      // addr[56:32] | type=2
  v8i g1;
  g1[0] = (int)((1u << 16) | (1u << 20) | (6u << 22) | (1u << 25)); // 2B, pad_en, interval, amount
  g1[1] = (int)((row_elems & 0xFFFFu) << 16);                // tensor_dim0 lo16
  g1[2] = (int)(((row_elems >> 16) & 0xFFFFu) | ((rows & 0xFFFFu) << 16));
  g1[3] = (int)(((rows >> 16) & 0xFFFFu) | ((row_elems & 0xFFFFu) << 16)); // tile_dim0
  g1[4] = (int)(rows & 0xFFFFu);                             // tile_dim1 (tile_dim2=0)
  g1[5] = (int)stride_elems;                                 // tensor_dim0_stride[31:0]
  g1[6] = 0;
  g1[7] = 0;
  v4i g2 = {0, 0, 0, 0};
  v4i g3 = {0, 0, 0, 0};
  v8i g4 = {0, 0, 0, 0, 0, 0, 0, 0};                         // unused (count==1)
  __builtin_amdgcn_tensor_load_to_lds(g0, g1, g2, g3, g4, 0);
}
#endif

// Stage `rows` x 256 bf16 (row-major, stride 256) into LDS with 520B pitch.
__device__ __forceinline__ void stage_rows_to_lds(char* smem, const u16* src, int rows) {
#if USE_TDM
  if ((threadIdx.x >> 5) == 0) {
    tdm_load_2d_bf16((u32)(uintptr_t)smem, src, (u32)rows, 256u, 256u);
    __builtin_amdgcn_s_wait_tensorcnt(0);
  }
  __syncthreads();
#else
  const u32 nwords = (u32)rows * 128u;
  for (u32 idx = threadIdx.x; idx < nwords; idx += blockDim.x) {
    const u32 row = idx >> 7, w = idx & 127u;
    *(u32*)(smem + row * LDSPITCH + (w << 2)) = ((const u32*)src)[idx];
  }
  __syncthreads();
#endif
}

// ---------------------------------------------------------------------------
// k0: proj f32 -> bf16
__global__ void k0_cvt_proj(const float* __restrict__ proj, u16* __restrict__ projb) {
  int i = (int)(blockIdx.x * blockDim.x + threadIdx.x);
  if (i < MMDIM * DDIM) projb[i] = f2bf(proj[i]);
}

// ---------------------------------------------------------------------------
// k1: X' = exp((X*0.25) @ proj^T - 0.5*||X*0.25||^2 - ln 16), optional mask.
// TRANS=false: out[b][row][m] (qs'); TRANS=true: out[b][m][row] (ks'^T, masked).
__device__ __forceinline__ void k1_araw(ARaw& ar, const float* Xrow, int d0, int hsel) {
  ((float4*)ar.f)[0] = *(const float4*)(Xrow + d0 + hsel);
  ((float4*)ar.f)[1] = *(const float4*)(Xrow + d0 + hsel + 4);
  ((float4*)ar.f)[2] = *(const float4*)(Xrow + d0 + 16 + hsel);
  ((float4*)ar.f)[3] = *(const float4*)(Xrow + d0 + 16 + hsel + 4);
}

template <bool TRANS>
__global__ __launch_bounds__(256) void k1_feat(const float* __restrict__ X,
                                               const u16* __restrict__ projb,
                                               u16* __restrict__ out,
                                               const int* __restrict__ vlens,
                                               int R) {
  extern __shared__ char smem[];                 // 256 rows x 520B (proj bf16)
  const int lane = (int)(threadIdx.x & 31u);
  const int wave = (int)(threadIdx.x >> 5u);
  const int bpb  = R >> 7;                       // blocks per batch (128 rows/block)
  const int b  = (int)blockIdx.x / bpb;
  const int rb = (int)blockIdx.x % bpb;
  const int r0 = rb * 128 + wave * 16;
  const int hsel = (lane & 16) ? 8 : 0;

  stage_rows_to_lds(smem, projb, MMDIM);

  const float* Xrow = X + ((size_t)b * (size_t)R + (size_t)(r0 + (lane & 15))) * DDIM;

  v8f acc[16];
#pragma unroll
  for (int t = 0; t < 16; ++t) acc[t] = vzero();

  float sq = 0.0f;
  ARaw ar;
  k1_araw(ar, Xrow, 0, hsel);                    // prefetch A for dstep 0

#pragma unroll 1
  for (int dstep = 0; dstep < 8; ++dstep) {
    Frag a;                                      // A: 16x32 bf16 per-lane layout
#pragma unroll
    for (int j = 0; j < 8; ++j) {                // pairs (0,1)..(14,15)
      float x0 = 0.25f * ar.f[2 * j];
      float x1 = 0.25f * ar.f[2 * j + 1];
      sq += x0 * x0 + x1 * x1;
      a.w[j] = pack2(x0, x1);
    }
    if (dstep < 7) k1_araw(ar, Xrow, (dstep + 1) * 32, hsel);  // hide behind wmma chain

    const u32 bofs = 2u * (u32)(dstep * 32 + hsel);
    Frag bf[2];
    lds_bfrag(bf[0], smem, (lane & 15), bofs);
#pragma unroll
    for (int nt = 0; nt < 16; ++nt) {            // double-buffered B frags
      if (nt < 15) lds_bfrag(bf[(nt + 1) & 1], smem, (nt + 1) * 16 + (lane & 15), bofs);
      acc[nt] = wmma_bf16(a, bf[nt & 1], acc[nt]);
    }
  }

  // per-row exponent bias: -0.5*||x~||^2 - ln(16); lane L holds row L&15 half-sum
  sq += __shfl_xor(sq, 16, 32);
  const float rt_self = -0.5f * sq - 2.7725887222397811f;
  float rt8[8];
#pragma unroll
  for (int r = 0; r < 8; ++r) rt8[r] = __shfl(rt_self, r + hsel, 32);

  if constexpr (!TRANS) {
    u16* ob = out + (size_t)b * (size_t)R * MMDIM;
#pragma unroll
    for (int nt = 0; nt < 16; ++nt) {
      const int col = nt * 16 + (lane & 15);
#pragma unroll
      for (int r = 0; r < 8; ++r) {
        const int row = r0 + r + hsel;
        ob[(size_t)row * MMDIM + col] = f2bf(__expf(acc[nt][r] + rt8[r]));
      }
    }
  } else {
    const int vlen = vlens[b];
    float keep8[8];
#pragma unroll
    for (int r = 0; r < 8; ++r) keep8[r] = (r0 + r + hsel < vlen) ? 1.0f : 0.0f;
    u16* ob = out + (size_t)b * MMDIM * (size_t)R;
#pragma unroll
    for (int nt = 0; nt < 16; ++nt) {
      const int m = nt * 16 + (lane & 15);
      uint4 d;
      d.x = pack2(__expf(acc[nt][0] + rt8[0]) * keep8[0], __expf(acc[nt][1] + rt8[1]) * keep8[1]);
      d.y = pack2(__expf(acc[nt][2] + rt8[2]) * keep8[2], __expf(acc[nt][3] + rt8[3]) * keep8[3]);
      d.z = pack2(__expf(acc[nt][4] + rt8[4]) * keep8[4], __expf(acc[nt][5] + rt8[5]) * keep8[5]);
      d.w = pack2(__expf(acc[nt][6] + rt8[6]) * keep8[6], __expf(acc[nt][7] + rt8[7]) * keep8[7]);
      *(uint4*)(ob + (size_t)m * (size_t)R + (size_t)(r0 + hsel)) = d;  // 8 contig k
    }
  }
}

// ---------------------------------------------------------------------------
// k2: buf1T[c][m] = sum_k C[k][c] * ks'[k][m];  C = [vs | 1 | 0-pad]
__device__ __forceinline__ void k2_araw(ARaw& ar, const float* vsb, int kk, int hsel,
                                        int c, bool cvalid, float cfill) {
#pragma unroll
  for (int j = 0; j < 8; ++j)
    ar.f[j] = cvalid ? vsb[(size_t)(kk + hsel + j) * DVDIM + c] : cfill;
#pragma unroll
  for (int j = 0; j < 8; ++j)
    ar.f[8 + j] = cvalid ? vsb[(size_t)(kk + 16 + hsel + j) * DVDIM + c] : cfill;
}

__global__ __launch_bounds__(256) void k2_buf1(const float* __restrict__ vs,
                                               const u16* __restrict__ kspT,
                                               float* __restrict__ buf1p) {
  const int lane = (int)(threadIdx.x & 31u);
  const int wave = (int)(threadIdx.x >> 5u);
  int id = (int)blockIdx.x;
  const int cblk  = id % NCBLK;  id /= NCBLK;
  const int split = id % NSPLIT; id /= NSPLIT;
  const int b = id;
  const int ct = cblk * 8 + wave;
  if (ct >= CTILES) return;

  const int hsel = (lane & 16) ? 8 : 0;
  const int c = ct * 16 + (lane & 15);
  const bool cvalid = (c < DVDIM);
  const float cfill = (c == DVDIM) ? 1.0f : 0.0f;
  const float* vsb = vs + (size_t)b * KKDIM * DVDIM;
  const u16* kb = kspT + (size_t)b * MMDIM * KKDIM;
  const int kbase0 = split * KCHUNK;

  v8f acc[16];
#pragma unroll
  for (int t = 0; t < 16; ++t) acc[t] = vzero();

  ARaw ar;
  k2_araw(ar, vsb, kbase0, hsel, c, cvalid, cfill);   // prefetch A for kstep 0

#pragma unroll 1
  for (int kstep = 0; kstep < KCHUNK / 32; ++kstep) {
    Frag a;                                      // A[c][k] = C[k][c]
#pragma unroll
    for (int j = 0; j < 8; ++j) a.w[j] = pack2(ar.f[2 * j], ar.f[2 * j + 1]);
    if (kstep < KCHUNK / 32 - 1)
      k2_araw(ar, vsb, kbase0 + (kstep + 1) * 32, hsel, c, cvalid, cfill);

    const u16* kbp = kb + (kbase0 + kstep * 32 + hsel);
    Frag bf[2];
    {
      const uint4* bp = (const uint4*)(kbp + (size_t)(lane & 15) * KKDIM);
      bf[0].q[0] = bp[0]; bf[0].q[1] = bp[2];
    }
#pragma unroll
    for (int mt = 0; mt < 16; ++mt) {            // double-buffered B frags
      if (mt < 15) {
        const uint4* bp = (const uint4*)(kbp + (size_t)((mt + 1) * 16 + (lane & 15)) * KKDIM);
        bf[(mt + 1) & 1].q[0] = bp[0];
        bf[(mt + 1) & 1].q[1] = bp[2];
      }
      acc[mt] = wmma_bf16(a, bf[mt & 1], acc[mt]);
    }
  }

  float* op = buf1p + ((size_t)split * BB + b) * (size_t)CPAD * MMDIM;
#pragma unroll
  for (int mt = 0; mt < 16; ++mt) {
    const int m = mt * 16 + (lane & 15);
#pragma unroll
    for (int r = 0; r < 8; ++r) {
      const int crow = ct * 16 + r + hsel;
      op[(size_t)crow * MMDIM + m] = acc[mt][r];
    }
  }
}

// k2b: reduce NSPLIT f32 partials -> bf16 buf1T (packed pair stores)
__global__ void k2b_reduce(const float* __restrict__ buf1p, u32* __restrict__ buf1T) {
  const size_t n2 = (size_t)BB * CPAD * MMDIM / 2;
  size_t i = (size_t)blockIdx.x * blockDim.x + threadIdx.x;
  if (i >= n2) return;
  const size_t stride = (size_t)BB * CPAD * MMDIM;
  const size_t e = i * 2;
  float s0 = 0.0f, s1 = 0.0f;
#pragma unroll
  for (int s = 0; s < NSPLIT; ++s) {
    s0 += buf1p[(size_t)s * stride + e];
    s1 += buf1p[(size_t)s * stride + e + 1];
  }
  buf1T[i] = pack2(s0, s1);
}

// ---------------------------------------------------------------------------
// k3: buf2 = qs' @ buf1; ctx = buf2[:, :256] / buf2[:, 256]
__global__ __launch_bounds__(256) void k3_out(const u16* __restrict__ qsp,
                                              const u16* __restrict__ buf1T,
                                              float* __restrict__ outp) {
  extern __shared__ char smem[];                 // 272 rows x 520B (buf1T bf16)
  const int lane = (int)(threadIdx.x & 31u);
  const int wave = (int)(threadIdx.x >> 5u);
  const int bpb = QQ >> 7;
  const int b  = (int)blockIdx.x / bpb;
  const int qb = (int)blockIdx.x % bpb;
  const int q0 = qb * 128 + wave * 16;
  const int hsel = (lane & 16) ? 8 : 0;

  stage_rows_to_lds(smem, buf1T + (size_t)b * CPAD * MMDIM, CPAD);

  const u16* qrow = qsp + ((size_t)b * QQ + (size_t)(q0 + (lane & 15))) * MMDIM;

  v8f acc[17];
#pragma unroll
  for (int t = 0; t < 17; ++t) acc[t] = vzero();

  Frag anext;
  anext.q[0] = *(const uint4*)(qrow + hsel);
  anext.q[1] = *(const uint4*)(qrow + 16 + hsel);

#pragma unroll 1
  for (int ms = 0; ms < 8; ++ms) {
    Frag a = anext;
    if (ms < 7) {
      const int m0 = (ms + 1) * 32;
      anext.q[0] = *(const uint4*)(qrow + m0 + hsel);
      anext.q[1] = *(const uint4*)(qrow + m0 + 16 + hsel);
    }
    const u32 bofs = 2u * (u32)(ms * 32 + hsel);
    Frag bf[2];
    lds_bfrag(bf[0], smem, (lane & 15), bofs);
#pragma unroll
    for (int ctile = 0; ctile < 17; ++ctile) {   // double-buffered B frags
      if (ctile < 16) lds_bfrag(bf[(ctile + 1) & 1], smem, (ctile + 1) * 16 + (lane & 15), bofs);
      acc[ctile] = wmma_bf16(a, bf[ctile & 1], acc[ctile]);
    }
  }

  // denominator = buf2[:,256] = tile 16, N=0 -> lane 0 (rows 0..7) / lane 16 (rows 8..15)
  float rden[8];
#pragma unroll
  for (int r = 0; r < 8; ++r) {
    const float den = __shfl(acc[16][r], lane & 16, 32);
    rden[r] = 1.0f / den;
  }

  float* ob = outp + (size_t)b * QQ * DVDIM;
#pragma unroll
  for (int ctile = 0; ctile < 16; ++ctile) {
    const int col = ctile * 16 + (lane & 15);
#pragma unroll
    for (int r = 0; r < 8; ++r) {
      const int row = q0 + r + hsel;
      ob[(size_t)row * DVDIM + col] = acc[ctile][r] * rden[r];
    }
  }
}

// ---------------------------------------------------------------------------
extern "C" void kernel_launch(void* const* d_in, const int* in_sizes, int n_in,
                              void* d_out, int out_size, void* d_ws, size_t ws_size,
                              hipStream_t stream) {
  const float* qs    = (const float*)d_in[0];
  const float* ks    = (const float*)d_in[1];
  const float* vs    = (const float*)d_in[2];
  const int*   vlens = (const int*)d_in[3];
  const float* proj  = (const float*)d_in[4];
  float* outp = (float*)d_out;
  (void)in_sizes; (void)n_in; (void)out_size; (void)ws_size;

  char* ws = (char*)d_ws;
  size_t off = 0;
  u16* projb = (u16*)(ws + off);   off += (size_t)MMDIM * DDIM * 2;                 // 128 KB
  u16* qsp   = (u16*)(ws + off);   off += (size_t)BB * QQ * MMDIM * 2;              // 32 MB
  u16* kspT  = (u16*)(ws + off);   off += (size_t)BB * MMDIM * KKDIM * 2;           // 32 MB
  float* buf1p = (float*)(ws + off); off += (size_t)NSPLIT * BB * CPAD * MMDIM * 4; // 17 MB
  u16* buf1T = (u16*)(ws + off);   off += (size_t)BB * CPAD * MMDIM * 2;            // 2.1 MB
  (void)off;

  k0_cvt_proj<<<dim3((MMDIM * DDIM + 255) / 256), dim3(256), 0, stream>>>(proj, projb);

  k1_feat<false><<<dim3(BB * (QQ / 128)), dim3(256), MMDIM * LDSPITCH, stream>>>(
      qs, projb, qsp, vlens, QQ);
  k1_feat<true><<<dim3(BB * (KKDIM / 128)), dim3(256), MMDIM * LDSPITCH, stream>>>(
      ks, projb, kspT, vlens, KKDIM);

  k2_buf1<<<dim3(BB * NSPLIT * NCBLK), dim3(256), 0, stream>>>(vs, kspT, buf1p);

  const u32 n2 = (u32)((size_t)BB * CPAD * MMDIM / 2);
  k2b_reduce<<<dim3((n2 + 255) / 256), dim3(256), 0, stream>>>(buf1p, (u32*)buf1T);

  k3_out<<<dim3(BB * (QQ / 128)), dim3(256), CPAD * LDSPITCH, stream>>>(qsp, buf1T, outp);
}
